// MSDeformAttn_89421219103225
// MI455X (gfx1250) — matrix operand order repaired and checked by reading, not compiled, not run on hardware
//
#include <hip/hip_runtime.h>
#include <hip/hip_bf16.h>
#include <math.h>

typedef __attribute__((ext_vector_type(16))) _Float16 v16h;
typedef __attribute__((ext_vector_type(8)))  float    v8f;

#define D_MODEL  256
#define N_HEADS  8
#define N_LEVELS 4
#define N_POINTS 4
#define C_HEAD   32
#define LEN_Q    37440
#define LEN_IN   37440

// -----------------------------------------------------------------------------
// C[M,N] = A[M,K] (f32 row-major) @ B[K,N] (f32 row-major) + bias[N]
// f16 WMMA, f32 accumulate. N,K compile-time; M%64==0, N%128==0, K%32==0.
// Block tile 64(M) x 128(N); 256 threads = 8 waves, wave w owns rows 0..63 of
// the tile and 16 columns (w*16..w*16+15) -> 4 accumulators, 4 WMMAs/K-step.
// LDS double-buffered: stage step k+1 while WMMAs consume step k.
// -----------------------------------------------------------------------------
template <int N, int K>
__launch_bounds__(256)
__global__ void gemm_wmma_f16(const float* __restrict__ A,
                              const float* __restrict__ Bw,
                              const float* __restrict__ bias,
                              float* __restrict__ C, int M) {
  __shared__ _Float16 As [2][64][40];    // [buf][m][k]  (+8 halves pad)
  __shared__ _Float16 BsT[2][128][40];   // [buf][n][k]  transposed B tile

  const int tid  = threadIdx.x;
  const int lane = tid & 31;
  const int w    = tid >> 5;            // wave 0..7 -> N sub-tile
  const int row0 = blockIdx.x * 64;
  const int col0 = blockIdx.y * 128;

  const int m_lane = lane & 15;         // A row / B col within 16-tile
  const int khalf  = lane >> 4;         // 0 or 1
  const int kbaseA = khalf * 8;         // unused; A K-halves are {0-15} / {16-31}

  v8f acc[4];
  #pragma unroll
  for (int mi = 0; mi < 4; ++mi) acc[mi] = v8f{0.f,0.f,0.f,0.f,0.f,0.f,0.f,0.f};

  // ---- staging: A 64x32 (8 floats/thr), B 32x128 -> transposed (16 floats/thr)
  auto stage = [&](int buf, int k0) {
    {
      const int i = tid * 8;
      const int r = i >> 5, c = i & 31;
      const float4* src =
          (const float4*)(A + (size_t)(row0 + r) * K + (k0 + c));
      float4 f0 = src[0], f1 = src[1];
      As[buf][r][c + 0] = (_Float16)f0.x;  As[buf][r][c + 1] = (_Float16)f0.y;
      As[buf][r][c + 2] = (_Float16)f0.z;  As[buf][r][c + 3] = (_Float16)f0.w;
      As[buf][r][c + 4] = (_Float16)f1.x;  As[buf][r][c + 5] = (_Float16)f1.y;
      As[buf][r][c + 6] = (_Float16)f1.z;  As[buf][r][c + 7] = (_Float16)f1.w;
    }
    {
      const int kr = tid >> 3;               // 0..31  (K row)
      const int c  = (tid & 7) * 16;         // 0..112 (N col, 16 per thread)
      const float4* src =
          (const float4*)(Bw + (size_t)(k0 + kr) * N + (col0 + c));
      #pragma unroll
      for (int jv = 0; jv < 4; ++jv) {
        float4 f = src[jv];
        BsT[buf][c + jv * 4 + 0][kr] = (_Float16)f.x;
        BsT[buf][c + jv * 4 + 1][kr] = (_Float16)f.y;
        BsT[buf][c + jv * 4 + 2][kr] = (_Float16)f.z;
        BsT[buf][c + jv * 4 + 3][kr] = (_Float16)f.w;
      }
    }
  };

  stage(0, 0);
  __syncthreads();

  #pragma unroll
  for (int kk = 0; kk < K / 32; ++kk) {
    const int cur = kk & 1;
    if (kk + 1 < K / 32) stage(cur ^ 1, (kk + 1) * 32);

    // B fragment: col n = w*16 + m_lane, K = khalf*16 + e (contiguous -> b128s)
    v16h b;
    #pragma unroll
    for (int e = 0; e < 16; ++e)
      b[e] = BsT[cur][w * 16 + m_lane][khalf * 16 + e];

    #pragma unroll
    for (int mi = 0; mi < 4; ++mi) {
      v16h a;
      #pragma unroll
      for (int e = 0; e < 8; ++e) {
        a[e]     = As[cur][mi * 16 + m_lane][khalf * 16 + e];
        a[e + 8] = As[cur][mi * 16 + m_lane][khalf * 16 + 8 + e];
      }
      acc[mi] = __builtin_amdgcn_wmma_f32_16x16x32_f16(
          false, a, false, b, (short)0, acc[mi], false, false);
    }
    __syncthreads();
  }

  // ---- epilogue: lane = N (lane&15), VGPR v -> M = v + 8*(lane>=16)
  const int n  = lane & 15;
  const int mb = (lane >> 4) * 8;
  const float bn = bias ? bias[col0 + w * 16 + n] : 0.f;
  #pragma unroll
  for (int mi = 0; mi < 4; ++mi) {
    float* crow = C + (size_t)(row0 + mi * 16 + mb) * N + (col0 + w * 16 + n);
    #pragma unroll
    for (int v = 0; v < 8; ++v) crow[(size_t)v * N] = acc[mi][v] + bn;
  }
}

// -----------------------------------------------------------------------------
// Deformable sampling: one wave per (query, head), lane = channel.
// Softmax over 16 (level,point) logits fused in (lane-uniform math).
// value lives in L2 (38 MB << 192 MB): per-corner reads are one 128B line.
// -----------------------------------------------------------------------------
__launch_bounds__(256)
__global__ void msda_sample(const float* __restrict__ value,     // [LEN_IN][256]
                            const float* __restrict__ off_raw,   // [LEN_Q][384]
                            const float* __restrict__ attn_raw,  // [LEN_Q][128]
                            const float* __restrict__ refp,      // [LEN_Q][4][3]
                            const int*   __restrict__ shapes,    // [4][3] (H,W,D)
                            const int*   __restrict__ lvl_start, // [4]
                            float* __restrict__ mid) {           // [LEN_Q][256]
  const int q = blockIdx.x;
  const int h = threadIdx.x >> 5;   // head = wave id
  const int c = threadIdx.x & 31;   // channel = lane

  const float* ar = attn_raw + (size_t)q * (N_HEADS * 16) + h * 16;
  float amax = -1e30f;
  #pragma unroll
  for (int i = 0; i < 16; ++i) amax = fmaxf(amax, ar[i]);
  float aw[16], asum = 0.f;
  #pragma unroll
  for (int i = 0; i < 16; ++i) { aw[i] = __expf(ar[i] - amax); asum += aw[i]; }
  const float inv = 1.f / asum;

  float acc = 0.f;
  #pragma unroll
  for (int l = 0; l < N_LEVELS; ++l) {
    const int H = shapes[l * 3 + 0];
    const int W = shapes[l * 3 + 1];
    const int D = shapes[l * 3 + 2];
    const int base = lvl_start[l];
    const float rx = refp[((size_t)q * N_LEVELS + l) * 3 + 0];
    const float ry = refp[((size_t)q * N_LEVELS + l) * 3 + 1];
    const float rz = refp[((size_t)q * N_LEVELS + l) * 3 + 2];
    #pragma unroll
    for (int p = 0; p < N_POINTS; ++p) {
      const float* o =
          off_raw + (size_t)q * 384 + ((h * N_LEVELS + l) * N_POINTS + p) * 3;
      // norm = (W, H, D); vol dims (Dz,Dy,Dx) = (H, W, D)
      const float lx = rx + o[0] / (float)W;   // x indexes Dx = D
      const float ly = ry + o[1] / (float)H;   // y indexes Dy = W
      const float lz = rz + o[2] / (float)D;   // z indexes Dz = H
      // grid_sample align_corners=False:  i = loc*Dim - 0.5
      const float ix = lx * (float)D - 0.5f;
      const float iy = ly * (float)W - 0.5f;
      const float iz = lz * (float)H - 0.5f;
      const float x0f = floorf(ix), y0f = floorf(iy), z0f = floorf(iz);
      const float fx = ix - x0f, fy = iy - y0f, fz = iz - z0f;
      const int x0 = (int)x0f, y0 = (int)y0f, z0 = (int)z0f;
      const float wa = inv * aw[l * N_POINTS + p];
      #pragma unroll
      for (int dz = 0; dz < 2; ++dz)
        #pragma unroll
        for (int dy = 0; dy < 2; ++dy)
          #pragma unroll
          for (int dx = 0; dx < 2; ++dx) {
            const int xc = x0 + dx, yc = y0 + dy, zc = z0 + dz;
            const bool valid = (xc >= 0) & (xc < D) & (yc >= 0) & (yc < W) &
                               (zc >= 0) & (zc < H);
            if (valid) {
              const float wgt = (dx ? fx : 1.f - fx) *
                                (dy ? fy : 1.f - fy) *
                                (dz ? fz : 1.f - fz);
              const int idx = (zc * W + yc) * D + xc;
              acc += wa * wgt *
                     value[(size_t)(base + idx) * D_MODEL + h * C_HEAD + c];
            }
          }
    }
  }
  mid[(size_t)q * D_MODEL + h * C_HEAD + c] = acc;
}

// -----------------------------------------------------------------------------
extern "C" void kernel_launch(void* const* d_in, const int* in_sizes, int n_in,
                              void* d_out, int out_size, void* d_ws, size_t ws_size,
                              hipStream_t stream) {
  const float* query   = (const float*)d_in[0];
  const float* refp    = (const float*)d_in[1];
  const float* in_flat = (const float*)d_in[2];
  const int*   shapes  = (const int*)d_in[3];
  const int*   lstart  = (const int*)d_in[4];
  const float* W_value = (const float*)d_in[5];
  const float* b_value = (const float*)d_in[6];
  const float* W_off   = (const float*)d_in[7];
  const float* b_off   = (const float*)d_in[8];
  const float* W_attn  = (const float*)d_in[9];
  const float* b_attn  = (const float*)d_in[10];
  const float* W_out   = (const float*)d_in[11];
  const float* b_out   = (const float*)d_in[12];

  // workspace partition
  char* ws = (char*)d_ws;
  float* value    = (float*)ws;  ws += (size_t)LEN_IN * 256 * sizeof(float);
  float* off_raw  = (float*)ws;  ws += (size_t)LEN_Q  * 384 * sizeof(float);
  float* attn_raw = (float*)ws;  ws += (size_t)LEN_Q  * 128 * sizeof(float);
  float* mid      = (float*)ws;

  const dim3 blk(256);

  // value = input_flatten @ W_value + b_value        (37440 x 256 x 256)
  gemm_wmma_f16<256, 256><<<dim3(LEN_IN / 64, 256 / 128), blk, 0, stream>>>(
      in_flat, W_value, b_value, value, LEN_IN);
  // off_raw = query @ W_off + b_off                  (37440 x 384 x 256)
  gemm_wmma_f16<384, 256><<<dim3(LEN_Q / 64, 384 / 128), blk, 0, stream>>>(
      query, W_off, b_off, off_raw, LEN_Q);
  // attn_raw = query @ W_attn + b_attn               (37440 x 128 x 256)
  gemm_wmma_f16<128, 256><<<dim3(LEN_Q / 64, 128 / 128), blk, 0, stream>>>(
      query, W_attn, b_attn, attn_raw, LEN_Q);
  // fused softmax + trilinear gather + weighted sum  -> mid (37440 x 256)
  msda_sample<<<dim3(LEN_Q), blk, 0, stream>>>(
      value, off_raw, attn_raw, refp, shapes, lstart, mid);
  // out = mid @ W_out + b_out                        (37440 x 256 x 256)
  gemm_wmma_f16<256, 256><<<dim3(LEN_Q / 64, 256 / 128), blk, 0, stream>>>(
      mid, W_out, b_out, (float*)d_out, LEN_Q);
}
